// Encoder_6648609374561
// MI455X (gfx1250) — matrix operand order repaired and checked by reading, not compiled, not run on hardware
//
#include <hip/hip_runtime.h>
#include <hip/hip_bf16.h>

// ---------------------------------------------------------------------------
// GRU encoder for MI455X (gfx1250, wave32, WMMA bf16 16x16x32, fp32 accum)
// B=64, T=2048, E=256, H=512, V=32000
//
// Persistent column-split scan: 16 WGs x 256 threads, weight slices LDS-
// resident (144KB of the 320KB WGP LDS), h/rh exchanged through L2, split
// arrive/wait grid barrier with input-side WMMAs hidden in the barrier shadow.
// ---------------------------------------------------------------------------

#define B_DIM 64
#define T_DIM 2048
#define E_DIM 256
#define H_DIM 512
#define NWG   16          // persistent workgroups; each owns 32 columns of H
#define SLICE 32          // H_DIM / NWG

typedef __bf16 bf16_t;
typedef __attribute__((ext_vector_type(16))) __bf16 v16bf;
typedef __attribute__((ext_vector_type(8)))  __bf16 v8bf;
typedef __attribute__((ext_vector_type(8)))  float  v8f;

// ---------------- WMMA fragment helpers (layouts per CDNA5 ISA 7.12.2) -----

// A-matrix 16x32 bf16: lane L<16 holds row L, K in {0..7, 16..23};
// lane L>=16 holds row L-16, K in {8..15, 24..31}. Two b128 loads.
__device__ __forceinline__ v16bf load_frag_a(const bf16_t* src, size_t ld,
                                             int row0, int k0, int lane) {
  const bf16_t* p = src + (size_t)(row0 + (lane & 15)) * ld
                        + (size_t)(k0 + ((lane >> 4) << 3));
  union { v16bf v; v8bf h[2]; } u;
  u.h[0] = *(const v8bf*)(p);        // K = base .. base+7
  u.h[1] = *(const v8bf*)(p + 16);   // K = base+16 .. base+23
  return u.v;
}

// B-matrix 32x16 bf16 (we compute h @ W^T, so B[k][n] = W[n][k]):
// lane holds column n = lane&15, K in {0..7, 16..23} (lane<16) or {8..15, 24..31}.
__device__ __forceinline__ v16bf load_frag_b(const bf16_t* w, size_t ld,
                                             int n0, int k0, int lane) {
  const bf16_t* p = w + (size_t)(n0 + (lane & 15)) * ld
                      + (size_t)(k0 + ((lane >> 4) << 3));
  union { v16bf v; v8bf h[2]; } u;
  u.h[0] = *(const v8bf*)(p);
  u.h[1] = *(const v8bf*)(p + 16);
  return u.v;
}

__device__ __forceinline__ v8f wmma_bf16(v16bf a, v16bf b, v8f c) {
  return __builtin_amdgcn_wmma_f32_16x16x32_bf16(
      /*neg_a=*/false, a, /*neg_b=*/false, b,
      /*c_mod=*/(short)0, c, /*reuse_a=*/false, /*reuse_b=*/false);
}

__device__ __forceinline__ float sigmoidf_dev(float x) {
  return 1.0f / (1.0f + __expf(-x));
}

// ---------------- split grid barrier (persistent kernel) -------------------
// arrive(): release this WG's writes and register arrival; returns the
// generation snapshot (valid in thread 0 only).
// wait():   spin on generation bump, then acquire (WGP$ invalidate) so the
// post-barrier loads see other WGPs' h/rh stores.
// Independent work placed between arrive() and wait() hides barrier latency.

__device__ __forceinline__ unsigned bar_arrive(unsigned* bar) {
  __threadfence();            // global_wb: publish rh/h stores device-wide
  __syncthreads();
  unsigned snap = 0u;
  if (threadIdx.x == 0) {
    snap = __hip_atomic_load(&bar[1], __ATOMIC_RELAXED,
                             __HIP_MEMORY_SCOPE_AGENT);
    unsigned arrived = __hip_atomic_fetch_add(&bar[0], 1u, __ATOMIC_ACQ_REL,
                                              __HIP_MEMORY_SCOPE_AGENT);
    if (arrived == NWG - 1u) {
      __hip_atomic_store(&bar[0], 0u, __ATOMIC_RELAXED,
                         __HIP_MEMORY_SCOPE_AGENT);
      __hip_atomic_fetch_add(&bar[1], 1u, __ATOMIC_RELEASE,
                             __HIP_MEMORY_SCOPE_AGENT);
    }
  }
  return snap;
}

__device__ __forceinline__ void bar_wait(unsigned* bar, unsigned snap) {
  if (threadIdx.x == 0) {
    while (__hip_atomic_load(&bar[1], __ATOMIC_ACQUIRE,
                             __HIP_MEMORY_SCOPE_AGENT) == snap) {
      __builtin_amdgcn_s_sleep(2);
    }
  }
  __syncthreads();
  __builtin_amdgcn_fence(__ATOMIC_ACQUIRE, "agent");  // invalidate WGP$
}

__device__ __forceinline__ void copy_to_lds(bf16_t* dst, const bf16_t* src,
                                            int nelem) {
  uint4* d = (uint4*)dst;
  const uint4* s = (const uint4*)src;
  int nv = nelem >> 3;   // 8 bf16 per uint4
  for (int i = threadIdx.x; i < nv; i += blockDim.x) d[i] = s[i];
}

// ---------------- prep kernels ---------------------------------------------

__global__ void cast_f32_bf16_kernel(const float* __restrict__ src,
                                     bf16_t* __restrict__ dst, int n) {
  int i = blockIdx.x * blockDim.x + threadIdx.x;
  if (i < n) dst[i] = (bf16_t)src[i];
}

__global__ void init_state_kernel(bf16_t* __restrict__ h_bf,
                                  unsigned* __restrict__ bar) {
  for (int i = threadIdx.x; i < B_DIM * H_DIM; i += blockDim.x)
    h_bf[i] = (bf16_t)0.0f;
  if (threadIdx.x == 0) { bar[0] = 0u; bar[1] = 0u; }
}

__global__ void lengths_kernel(const int* __restrict__ X,
                               int* __restrict__ lengths) {
  __shared__ int red[256];
  int b = blockIdx.x;
  int cnt = 0;
  for (int t = threadIdx.x; t < T_DIM; t += blockDim.x)
    cnt += (X[(size_t)b * T_DIM + t] != 0) ? 1 : 0;
  red[threadIdx.x] = cnt;
  __syncthreads();
  for (int s = 128; s > 0; s >>= 1) {
    if (threadIdx.x < (unsigned)s) red[threadIdx.x] += red[threadIdx.x + s];
    __syncthreads();
  }
  if (threadIdx.x == 0) lengths[b] = red[0];
}

__global__ void embed_kernel(const int* __restrict__ X,
                             const float* __restrict__ emb,
                             bf16_t* __restrict__ xe, long total) {
  long stride = (long)gridDim.x * blockDim.x;
  for (long i = (long)blockIdx.x * blockDim.x + threadIdx.x; i < total;
       i += stride) {
    long bt = i >> 8;          // E_DIM == 256
    int  e  = (int)(i & 255);
    int tok = X[bt];
    xe[i] = (bf16_t)emb[(size_t)tok * E_DIM + e];
  }
}

// ---------------- persistent GRU scan kernel -------------------------------
// Wave w computes output tile (mt = w&3, nt = w>>2) of its WG's [64 x 32]
// column slice. Input-side (Xemb) WMMAs run in the barrier shadow.

__launch_bounds__(256, 1)
__global__ void gru_scan_kernel(
    bf16_t* __restrict__ h_bf,          // [64][512] bf16 shared state
    bf16_t* __restrict__ rh_bf,         // [64][512] bf16 r*h staging
    const bf16_t* __restrict__ Xemb,    // [64][2048][256] bf16
    const bf16_t* __restrict__ Ur,      // [512][512] bf16
    const bf16_t* __restrict__ Uz,
    const bf16_t* __restrict__ Whh,
    const bf16_t* __restrict__ Wr,      // [512][256] bf16
    const bf16_t* __restrict__ Wz,
    const bf16_t* __restrict__ Wxh,
    const bf16_t* __restrict__ Vw,      // [512][512] bf16
    const float* __restrict__ Wr_b, const float* __restrict__ Ur_b,
    const float* __restrict__ Wz_b, const float* __restrict__ Uz_b,
    const float* __restrict__ Wxh_b, const float* __restrict__ Whh_b,
    const float* __restrict__ V_b,
    const int* __restrict__ lengths,
    unsigned* __restrict__ bar,
    float* __restrict__ out)            // [64][512] fp32
{
  extern __shared__ bf16_t smem[];
  bf16_t* ldsUr  = smem;                       // [32][512]
  bf16_t* ldsUz  = ldsUr  + SLICE * H_DIM;
  bf16_t* ldsWhh = ldsUz  + SLICE * H_DIM;
  bf16_t* ldsWr  = ldsWhh + SLICE * H_DIM;     // [32][256]
  bf16_t* ldsWz  = ldsWr  + SLICE * E_DIM;
  bf16_t* ldsWxh = ldsWz  + SLICE * E_DIM;

  const int g    = blockIdx.x;                 // column-slice owner
  const int lane = threadIdx.x & 31;
  const int wave = threadIdx.x >> 5;
  const int mt   = wave & 3;                   // M tile: rows mt*16..+15
  const int nt   = wave >> 2;                  // N tile within slice (0..1)
  const int rlo  = (lane >> 4) << 3;           // C-frag row offset (0 or 8)
  const int colg = g * SLICE + nt * 16 + (lane & 15); // global H column
  const size_t xld = (size_t)T_DIM * E_DIM;    // Xemb batch-row stride

  // Stage this WG's weight slices into LDS once (reused 2048x).
  copy_to_lds(ldsUr,  Ur  + (size_t)g * SLICE * H_DIM, SLICE * H_DIM);
  copy_to_lds(ldsUz,  Uz  + (size_t)g * SLICE * H_DIM, SLICE * H_DIM);
  copy_to_lds(ldsWhh, Whh + (size_t)g * SLICE * H_DIM, SLICE * H_DIM);
  copy_to_lds(ldsWr,  Wr  + (size_t)g * SLICE * E_DIM, SLICE * E_DIM);
  copy_to_lds(ldsWz,  Wz  + (size_t)g * SLICE * E_DIM, SLICE * E_DIM);
  copy_to_lds(ldsWxh, Wxh + (size_t)g * SLICE * E_DIM, SLICE * E_DIM);
  __syncthreads();

  // Per-thread constants.
  const float bias_r = Wr_b[colg]  + Ur_b[colg];
  const float bias_z = Wz_b[colg]  + Uz_b[colg];
  const float bias_u = Wxh_b[colg] + Whh_b[colg];
  const float bias_o = V_b[colg];
  int lenr[8];
#pragma unroll
  for (int v = 0; v < 8; ++v) lenr[v] = lengths[mt * 16 + rlo + v];

  v8f h_reg;                                   // fp32 hidden state (own tile)
#pragma unroll
  for (int v = 0; v < 8; ++v) h_reg[v] = 0.0f;

  // Pre-compute input-side r/z contributions for t=0 (no dependencies).
  v8f xr_acc, xz_acc;
#pragma unroll
  for (int v = 0; v < 8; ++v) { xr_acc[v] = bias_r; xz_acc[v] = bias_z; }
#pragma unroll
  for (int kt = 0; kt < E_DIM / 32; ++kt) {
    v16bf a  = load_frag_a(Xemb, xld, mt * 16, kt * 32, lane);
    v16bf br = load_frag_b(ldsWr, E_DIM, nt * 16, kt * 32, lane);
    v16bf bz = load_frag_b(ldsWz, E_DIM, nt * 16, kt * 32, lane);
    xr_acc = wmma_bf16(a, br, xr_acc);
    xz_acc = wmma_bf16(a, bz, xz_acc);
  }

  for (int t = 0; t < T_DIM; ++t) {
    const bf16_t* xbase = Xemb + (size_t)t * E_DIM;

    // ---- Phase A (critical): recurrent r/z terms on current h ------------
    v8f r_acc = xr_acc, z_acc = xz_acc;
#pragma unroll
    for (int kt = 0; kt < H_DIM / 32; ++kt) {
      v16bf a  = load_frag_a(h_bf, H_DIM, mt * 16, kt * 32, lane);
      v16bf br = load_frag_b(ldsUr, H_DIM, nt * 16, kt * 32, lane);
      v16bf bz = load_frag_b(ldsUz, H_DIM, nt * 16, kt * 32, lane);
      r_acc = wmma_bf16(a, br, r_acc);
      z_acc = wmma_bf16(a, bz, z_acc);
    }
#pragma unroll
    for (int v = 0; v < 8; ++v) {
      float r  = sigmoidf_dev(r_acc[v]);
      int  row = mt * 16 + rlo + v;
      rh_bf[(size_t)row * H_DIM + colg] = (bf16_t)(r * h_reg[v]);
      z_acc[v] = sigmoidf_dev(z_acc[v]);       // keep z in registers
    }
    unsigned snap = bar_arrive(bar);

    // ---- Barrier shadow: input-side candidate term (independent) ---------
    v8f u_acc;
#pragma unroll
    for (int v = 0; v < 8; ++v) u_acc[v] = bias_u;
#pragma unroll
    for (int kt = 0; kt < E_DIM / 32; ++kt) {
      v16bf a = load_frag_a(xbase, xld, mt * 16, kt * 32, lane);
      v16bf b = load_frag_b(ldsWxh, E_DIM, nt * 16, kt * 32, lane);
      u_acc = wmma_bf16(a, b, u_acc);
    }
    bar_wait(bar, snap);                       // rh now globally visible

    // ---- Phase B (critical): candidate + state update --------------------
#pragma unroll
    for (int kt = 0; kt < H_DIM / 32; ++kt) {
      v16bf a = load_frag_a(rh_bf, H_DIM, mt * 16, kt * 32, lane);
      v16bf b = load_frag_b(ldsWhh, H_DIM, nt * 16, kt * 32, lane);
      u_acc = wmma_bf16(a, b, u_acc);
    }
#pragma unroll
    for (int v = 0; v < 8; ++v) {
      float upd = tanhf(u_acc[v]);
      float z   = z_acc[v];
      float hn  = z * h_reg[v] + (1.0f - z) * upd;
      if (t < lenr[v]) h_reg[v] = hn;          // length mask (freeze past end)
      int row = mt * 16 + rlo + v;
      h_bf[(size_t)row * H_DIM + colg] = (bf16_t)h_reg[v];
    }
    snap = bar_arrive(bar);

    // ---- Barrier shadow: input-side r/z terms for t+1 (independent) ------
    if (t + 1 < T_DIM) {
      const bf16_t* xnext = Xemb + (size_t)(t + 1) * E_DIM;
#pragma unroll
      for (int v = 0; v < 8; ++v) { xr_acc[v] = bias_r; xz_acc[v] = bias_z; }
#pragma unroll
      for (int kt = 0; kt < E_DIM / 32; ++kt) {
        v16bf a  = load_frag_a(xnext, xld, mt * 16, kt * 32, lane);
        v16bf br = load_frag_b(ldsWr, E_DIM, nt * 16, kt * 32, lane);
        v16bf bz = load_frag_b(ldsWz, E_DIM, nt * 16, kt * 32, lane);
        xr_acc = wmma_bf16(a, br, xr_acc);
        xz_acc = wmma_bf16(a, bz, xz_acc);
      }
    }
    bar_wait(bar, snap);                       // h(t+1) now globally visible
  }

  // ---------------- final projection: out = tanh(hT @ V_w^T + V_b) --------
  v8f o_acc;
#pragma unroll
  for (int v = 0; v < 8; ++v) o_acc[v] = bias_o;
#pragma unroll
  for (int kt = 0; kt < H_DIM / 32; ++kt) {
    v16bf a = load_frag_a(h_bf, H_DIM, mt * 16, kt * 32, lane);
    v16bf b = load_frag_b(Vw, H_DIM, g * SLICE + nt * 16, kt * 32, lane);
    o_acc = wmma_bf16(a, b, o_acc);
  }
#pragma unroll
  for (int v = 0; v < 8; ++v) {
    int row = mt * 16 + rlo + v;
    out[(size_t)row * H_DIM + colg] = tanhf(o_acc[v]);
  }
}

// ---------------------------------------------------------------------------
extern "C" void kernel_launch(void* const* d_in, const int* in_sizes, int n_in,
                              void* d_out, int out_size, void* d_ws,
                              size_t ws_size, hipStream_t stream) {
  (void)in_sizes; (void)n_in; (void)out_size; (void)ws_size;

  const int*   X     = (const int*)d_in[0];
  const float* emb   = (const float*)d_in[1];
  const float* Wr_w  = (const float*)d_in[2];
  const float* Wr_b  = (const float*)d_in[3];
  const float* Ur_w  = (const float*)d_in[4];
  const float* Ur_b  = (const float*)d_in[5];
  const float* Wz_w  = (const float*)d_in[6];
  const float* Wz_b  = (const float*)d_in[7];
  const float* Uz_w  = (const float*)d_in[8];
  const float* Uz_b  = (const float*)d_in[9];
  const float* Wxh_w = (const float*)d_in[10];
  const float* Wxh_b = (const float*)d_in[11];
  const float* Whh_w = (const float*)d_in[12];
  const float* Whh_b = (const float*)d_in[13];
  const float* V_w   = (const float*)d_in[14];
  const float* V_b   = (const float*)d_in[15];
  float* out = (float*)d_out;

  // Workspace layout (256B aligned slots).
  char* ws = (char*)d_ws;
  size_t off = 0;
  auto take = [&](size_t bytes) -> char* {
    char* p = ws + off;
    off = (off + bytes + 255) & ~(size_t)255;
    return p;
  };
  bf16_t* ur_bf  = (bf16_t*)take((size_t)H_DIM * H_DIM * 2);
  bf16_t* uz_bf  = (bf16_t*)take((size_t)H_DIM * H_DIM * 2);
  bf16_t* whh_bf = (bf16_t*)take((size_t)H_DIM * H_DIM * 2);
  bf16_t* vw_bf  = (bf16_t*)take((size_t)H_DIM * H_DIM * 2);
  bf16_t* wr_bf  = (bf16_t*)take((size_t)H_DIM * E_DIM * 2);
  bf16_t* wz_bf  = (bf16_t*)take((size_t)H_DIM * E_DIM * 2);
  bf16_t* wxh_bf = (bf16_t*)take((size_t)H_DIM * E_DIM * 2);
  bf16_t* h_bf   = (bf16_t*)take((size_t)B_DIM * H_DIM * 2);
  bf16_t* rh_bf  = (bf16_t*)take((size_t)B_DIM * H_DIM * 2);
  int*    lens   = (int*)take(B_DIM * sizeof(int));
  unsigned* bar  = (unsigned*)take(256);
  bf16_t* xemb   = (bf16_t*)take((size_t)B_DIM * T_DIM * E_DIM * 2);

  // Prep: weight casts to bf16.
  const int nHH = H_DIM * H_DIM, nHE = H_DIM * E_DIM;
  cast_f32_bf16_kernel<<<(nHH + 255) / 256, 256, 0, stream>>>(Ur_w,  ur_bf,  nHH);
  cast_f32_bf16_kernel<<<(nHH + 255) / 256, 256, 0, stream>>>(Uz_w,  uz_bf,  nHH);
  cast_f32_bf16_kernel<<<(nHH + 255) / 256, 256, 0, stream>>>(Whh_w, whh_bf, nHH);
  cast_f32_bf16_kernel<<<(nHH + 255) / 256, 256, 0, stream>>>(V_w,   vw_bf,  nHH);
  cast_f32_bf16_kernel<<<(nHE + 255) / 256, 256, 0, stream>>>(Wr_w,  wr_bf,  nHE);
  cast_f32_bf16_kernel<<<(nHE + 255) / 256, 256, 0, stream>>>(Wz_w,  wz_bf,  nHE);
  cast_f32_bf16_kernel<<<(nHE + 255) / 256, 256, 0, stream>>>(Wxh_w, wxh_bf, nHE);

  init_state_kernel<<<1, 256, 0, stream>>>(h_bf, bar);
  lengths_kernel<<<B_DIM, 256, 0, stream>>>(X, lens);

  const long total_emb = (long)B_DIM * T_DIM * E_DIM;
  embed_kernel<<<4096, 256, 0, stream>>>(X, emb, xemb, total_emb);

  // Persistent scan: 16 WGs, 256 threads, 144KB dynamic LDS (CDNA5: 320KB/WGP).
  const size_t lds_bytes =
      (size_t)(3 * SLICE * H_DIM + 3 * SLICE * E_DIM) * sizeof(bf16_t);
  hipFuncSetAttribute((const void*)gru_scan_kernel,
                      hipFuncAttributeMaxDynamicSharedMemorySize,
                      (int)lds_bytes);
  gru_scan_kernel<<<NWG, 256, lds_bytes, stream>>>(
      h_bf, rh_bf, xemb, ur_bf, uz_bf, whh_bf, wr_bf, wz_bf, wxh_bf, vw_bf,
      Wr_b, Ur_b, Wz_b, Uz_b, Wxh_b, Whh_b, V_b, lens, bar, out);
}
